// LexionAdapter_65283502899540
// MI455X (gfx1250) — matrix operand order repaired and checked by reading, not compiled, not run on hardware
//
#include <hip/hip_runtime.h>
#include <hip/hip_bf16.h>
#include <stdint.h>

// ---------------------------------------------------------------------------
// Types for WMMA
// ---------------------------------------------------------------------------
typedef __attribute__((ext_vector_type(16))) __bf16     v16bf;
typedef __attribute__((ext_vector_type(8)))  float      v8f;

__device__ __forceinline__ uint16_t f32_to_bf16(float f) {
  uint32_t u = __float_as_uint(f);
  uint32_t r = u + 0x7FFFu + ((u >> 16) & 1u);   // round-to-nearest-even
  return (uint16_t)(r >> 16);
}
__device__ __forceinline__ float bf16_lo(uint32_t u) {
  return __uint_as_float(u << 16);
}
__device__ __forceinline__ float bf16_hi(uint32_t u) {
  return __uint_as_float(u & 0xFFFF0000u);
}
// branchless tanh: 1 - 2/(e^{2x}+1); saturates to +/-1, no exec-mask branches
__device__ __forceinline__ float fast_tanh(float x) {
  float e = __expf(2.0f * x);
  float r = __builtin_amdgcn_rcpf(e + 1.0f);
  return __builtin_fmaf(-2.0f, r, 1.0f);
}

// ---------------------------------------------------------------------------
// Pack weights: fp32 [K][N] row-major  ->  bf16 [N][Kpad] (transposed, padded)
// ---------------------------------------------------------------------------
__global__ void pack_transpose_bf16(const float* __restrict__ W,
                                    uint16_t* __restrict__ out,
                                    int K, int N, int Kpad) {
  int idx = blockIdx.x * blockDim.x + threadIdx.x;
  int total = N * Kpad;
  if (idx >= total) return;
  int n = idx / Kpad;
  int k = idx - n * Kpad;
  float v = (k < K) ? W[(size_t)k * N + n] : 0.0f;
  out[idx] = f32_to_bf16(v);
}

// ---------------------------------------------------------------------------
// Pack activations: fp32 [M][K] -> bf16 [M][Kpad] zero-padded (coalesced)
// ---------------------------------------------------------------------------
__global__ void pack_rows_bf16(const float* __restrict__ A,
                               uint16_t* __restrict__ out,
                               int K, int Kpad, int total) {
  int idx = blockIdx.x * blockDim.x + threadIdx.x;
  if (idx >= total) return;
  int m = idx / Kpad;
  int k = idx - m * Kpad;
  float v = (k < K) ? A[(size_t)m * K + k] : 0.0f;
  out[idx] = f32_to_bf16(v);
}

// ---------------------------------------------------------------------------
// WMMA fragment loads from LDS (CDNA5 ISA 7.12.2 16-bit layouts).
//   A 16x32 : lanes 0-15 hold M=lane, K {0..7,16..23}; lanes 16-31: K {8..15,24..31}
//   B 32x16 : lanes 0-15 hold N=lane, K {0..7,16..23}; lanes 16-31: K {8..15,24..31}
// Rows are 80B-aligned so each fragment is two ds_load_b128.
// ---------------------------------------------------------------------------
__device__ __forceinline__ v16bf load_a_frag(const uint16_t* rowp, int lane) {
  union { uint4 q[2]; v16bf bf; } u;
  const int klo = (lane < 16) ? 0 : 8;
  u.q[0] = *(const uint4*)(rowp + klo);
  u.q[1] = *(const uint4*)(rowp + klo + 16);
  return u.bf;
}
__device__ __forceinline__ v16bf load_b_frag(const uint16_t* rowp, int lane) {
  union { uint4 q[2]; v16bf bf; } u;
  const int ko = (lane < 16) ? 0 : 8;
  u.q[0] = *(const uint4*)(rowp + ko);
  u.q[1] = *(const uint4*)(rowp + ko + 16);
  return u.bf;
}

// CDNA5 async DMA global->LDS, 16B per lane, ASYNCcnt-tracked.
// INST_OFFSET is added to BOTH the global and the LDS address (ISA 10.1 / 15.18.3),
// so one base VGPR pair serves both 16B chunks of a 32B row segment.
__device__ __forceinline__ void async_copy_32B(uint32_t lds_addr, const void* gsrc) {
  uint64_t ga = (uint64_t)(size_t)gsrc;
  asm volatile("global_load_async_to_lds_b128 %0, %1, off"
               :: "v"(lds_addr), "v"(ga) : "memory");
  asm volatile("global_load_async_to_lds_b128 %0, %1, off offset:16"
               :: "v"(lds_addr), "v"(ga) : "memory");
}
__device__ __forceinline__ void wait_async_zero() {
  asm volatile("s_wait_asynccnt 0x0" ::: "memory");
}

// ---------------------------------------------------------------------------
// Tiled bf16 WMMA GEMM:  C[M x N] = A[M x Kpad] * Bt^T  (+bias, +activation)
//   Bt pre-packed bf16 [N][Kpad].  AF32: A is fp32 (converted during stage-in,
//   requires lda == Kpad, no tail); else A is bf16 (already padded).
//   EPI : 0 = tanh(x+bias)->bf16 ; 1 = x+bias->bf16 ; 2 = x->f32
// BM=BN=128, BK=32, 256 threads = 8 waves (4 along M x 2 along N),
// each wave: 32x64 patch = 2x4 tiles of v_wmma_f32_16x16x32_bf16.
// ---------------------------------------------------------------------------
#define BM 128
#define BN 128
#define BK 32
#define LDSS 40   // ushort stride: 80B rows -> every 16B chunk aligned for b128

template <bool AF32, int EPI>
__global__ __launch_bounds__(256) void gemm_bf16_wmma(
    const void* __restrict__ Aptr, const uint16_t* __restrict__ Bt,
    const float* __restrict__ bias, void* __restrict__ Cptr,
    int lda, int Kpad, int ldc) {
  __shared__ uint16_t sA[BM * LDSS];
  __shared__ uint16_t sB[BN * LDSS];

  const int tid  = threadIdx.x;
  const int lane = tid & 31;
  const int wave = tid >> 5;
  const int wm   = wave & 3;   // wave row    (4 x 32 rows)
  const int wn   = wave >> 2;  // wave column (2 x 64 cols)

  const int m0 = blockIdx.x * BM;
  const int n0 = blockIdx.y * BN;

  v8f acc[2][4] = {};

  const int r  = tid >> 1;          // 0..127 : tile row staged by this thread
  const int c0 = (tid & 1) * 16;    // 0 / 16 : 16-element chunk
  uint16_t* sAdst = &sA[r * LDSS + c0];
  uint16_t* sBdst = &sB[r * LDSS + c0];
  // LDS flat address low 32 bits == LDS byte offset (aperture layout)
  const uint32_t sAaddr = (uint32_t)(size_t)(void*)sAdst;
  const uint32_t sBaddr = (uint32_t)(size_t)(void*)sBdst;

  const int ksteps = Kpad / BK;
  for (int kt = 0; kt < ksteps; ++kt) {
    const int kbase = kt * BK;
    // ---- stage A tile into LDS ----
    if (AF32) {
      const float4* af =
          (const float4*)((const float*)Aptr + (size_t)(m0 + r) * lda + kbase + c0);
      float4 f0 = af[0], f1 = af[1], f2 = af[2], f3 = af[3];
      union { uint32_t u[8]; uint4 q[2]; } pk;
      pk.u[0] = (uint32_t)f32_to_bf16(f0.x) | ((uint32_t)f32_to_bf16(f0.y) << 16);
      pk.u[1] = (uint32_t)f32_to_bf16(f0.z) | ((uint32_t)f32_to_bf16(f0.w) << 16);
      pk.u[2] = (uint32_t)f32_to_bf16(f1.x) | ((uint32_t)f32_to_bf16(f1.y) << 16);
      pk.u[3] = (uint32_t)f32_to_bf16(f1.z) | ((uint32_t)f32_to_bf16(f1.w) << 16);
      pk.u[4] = (uint32_t)f32_to_bf16(f2.x) | ((uint32_t)f32_to_bf16(f2.y) << 16);
      pk.u[5] = (uint32_t)f32_to_bf16(f2.z) | ((uint32_t)f32_to_bf16(f2.w) << 16);
      pk.u[6] = (uint32_t)f32_to_bf16(f3.x) | ((uint32_t)f32_to_bf16(f3.y) << 16);
      pk.u[7] = (uint32_t)f32_to_bf16(f3.z) | ((uint32_t)f32_to_bf16(f3.w) << 16);
      ((uint4*)sAdst)[0] = pk.q[0];
      ((uint4*)sAdst)[1] = pk.q[1];
    } else {
      // async DMA: no VGPR round-trip, tracked by ASYNCcnt
      async_copy_32B(sAaddr,
                     (const uint16_t*)Aptr + (size_t)(m0 + r) * lda + kbase + c0);
    }
    // ---- stage B tile (bf16 [N][Kpad] row-major) via async DMA ----
    async_copy_32B(sBaddr, Bt + (size_t)(n0 + r) * Kpad + kbase + c0);

    // prefetch next K tile while WMMAs run (global_prefetch_b8)
    if (kt + 1 < ksteps) {
      if (AF32)
        __builtin_prefetch((const float*)Aptr + (size_t)(m0 + r) * lda + kbase + BK + c0, 0, 3);
      else
        __builtin_prefetch((const uint16_t*)Aptr + (size_t)(m0 + r) * lda + kbase + BK + c0, 0, 3);
      __builtin_prefetch(Bt + (size_t)(n0 + r) * Kpad + kbase + BK + c0, 0, 3);
    }
    wait_async_zero();
    __syncthreads();

    // ---- fragments + 2x4 WMMA ----
    v16bf afr[2], bfr[4];
#pragma unroll
    for (int mt = 0; mt < 2; ++mt)
      afr[mt] = load_a_frag(&sA[(wm * 32 + mt * 16 + (lane & 15)) * LDSS], lane);
#pragma unroll
    for (int nt = 0; nt < 4; ++nt)
      bfr[nt] = load_b_frag(&sB[(wn * 64 + nt * 16 + (lane & 15)) * LDSS], lane);
#pragma unroll
    for (int mt = 0; mt < 2; ++mt)
#pragma unroll
      for (int nt = 0; nt < 4; ++nt)
        acc[mt][nt] = __builtin_amdgcn_wmma_f32_16x16x32_bf16(
            false, afr[mt], false, bfr[nt], (short)0, acc[mt][nt], false, false);
    __syncthreads();
  }

  // ---- epilogue: C layout = VGPR r -> row r (lanes<16) / r+8 (lanes>=16) ----
#pragma unroll
  for (int mt = 0; mt < 2; ++mt) {
    const int mbase = m0 + wm * 32 + mt * 16 + ((lane < 16) ? 0 : 8);
#pragma unroll
    for (int nt = 0; nt < 4; ++nt) {
      const int n = n0 + wn * 64 + nt * 16 + (lane & 15);
      const float bv = (EPI == 2) ? 0.0f : bias[n];
#pragma unroll
      for (int rr = 0; rr < 8; ++rr) {
        float v = acc[mt][nt][rr] + bv;
        size_t off = (size_t)(mbase + rr) * ldc + n;
        if (EPI == 0) {
          ((uint16_t*)Cptr)[off] = f32_to_bf16(fast_tanh(v));
        } else if (EPI == 1) {
          ((uint16_t*)Cptr)[off] = f32_to_bf16(v);
        } else {
          ((float*)Cptr)[off] = v;
        }
      }
    }
  }
}

// ---------------------------------------------------------------------------
// Fused bilinear attention + softmax + weighted sum + residual + LayerNorm.
// One wave32 per token; H=768 -> 6 chunks of 4 contiguous elements per lane.
// ---------------------------------------------------------------------------
#define HH 768
#define NW 8

__global__ __launch_bounds__(256) void fuse_attn_ln(
    const float* __restrict__ Q, const uint16_t* __restrict__ WO,
    const float* __restrict__ LO, const int* __restrict__ mask,
    const float* __restrict__ ln_g, const float* __restrict__ ln_b,
    float* __restrict__ out) {
  const int lane = threadIdx.x & 31;
  const int tok  = blockIdx.x * 8 + (threadIdx.x >> 5);
  const int hofs = 4 * lane;                 // chunk base within each 128 block

  const float*    q  = Q  + (size_t)tok * HH;
  const uint16_t* wo = WO + (size_t)tok * NW * HH;

  float4 qv[6];
#pragma unroll
  for (int j = 0; j < 6; ++j) qv[j] = *(const float4*)(q + 128 * j + hofs);

  // bilinear scores, wave-reduced
  float sc[NW];
#pragma unroll
  for (int w = 0; w < NW; ++w) {
    float p = 0.0f;
#pragma unroll
    for (int j = 0; j < 6; ++j) {
      const uint32_t* wp = (const uint32_t*)(wo + w * HH + 128 * j + hofs);
      uint32_t a = wp[0], b = wp[1];
      p += qv[j].x * bf16_lo(a) + qv[j].y * bf16_hi(a) +
           qv[j].z * bf16_lo(b) + qv[j].w * bf16_hi(b);
    }
#pragma unroll
    for (int off = 16; off > 0; off >>= 1) p += __shfl_xor(p, off, 32);
    sc[w] = p;
  }

  // mask + softmax over W=8
#pragma unroll
  for (int w = 0; w < NW; ++w)
    sc[w] += (1.0f - (float)mask[(size_t)tok * NW + w]) * -10000.0f;
  float mx = sc[0];
#pragma unroll
  for (int w = 1; w < NW; ++w) mx = fmaxf(mx, sc[w]);
  float ssum = 0.0f;
  float alpha[NW];
#pragma unroll
  for (int w = 0; w < NW; ++w) { alpha[w] = __expf(sc[w] - mx); ssum += alpha[w]; }
  const float rs = 1.0f / ssum;
#pragma unroll
  for (int w = 0; w < NW; ++w) alpha[w] *= rs;

  // weighted sum + residual
  float4 y[6];
  float s = 0.0f;
#pragma unroll
  for (int j = 0; j < 6; ++j) {
    float4 a = *(const float4*)(LO + (size_t)tok * HH + 128 * j + hofs);
#pragma unroll
    for (int w = 0; w < NW; ++w) {
      const uint32_t* wp = (const uint32_t*)(wo + w * HH + 128 * j + hofs);
      uint32_t u0 = wp[0], u1 = wp[1];
      a.x += alpha[w] * bf16_lo(u0);
      a.y += alpha[w] * bf16_hi(u0);
      a.z += alpha[w] * bf16_lo(u1);
      a.w += alpha[w] * bf16_hi(u1);
    }
    y[j] = a;
    s += a.x + a.y + a.z + a.w;
  }
#pragma unroll
  for (int off = 16; off > 0; off >>= 1) s += __shfl_xor(s, off, 32);
  const float mu = s * (1.0f / (float)HH);
  float ss = 0.0f;
#pragma unroll
  for (int j = 0; j < 6; ++j) {
    float dx = y[j].x - mu, dy = y[j].y - mu, dz = y[j].z - mu, dw = y[j].w - mu;
    ss += dx * dx + dy * dy + dz * dz + dw * dw;
  }
#pragma unroll
  for (int off = 16; off > 0; off >>= 1) ss += __shfl_xor(ss, off, 32);
  const float inv = rsqrtf(ss * (1.0f / (float)HH) + 1e-12f);

#pragma unroll
  for (int j = 0; j < 6; ++j) {
    const int h = 128 * j + hofs;
    float4 g4 = *(const float4*)(ln_g + h);
    float4 b4 = *(const float4*)(ln_b + h);
    float4 o;
    o.x = (y[j].x - mu) * inv * g4.x + b4.x;
    o.y = (y[j].y - mu) * inv * g4.y + b4.y;
    o.z = (y[j].z - mu) * inv * g4.z + b4.z;
    o.w = (y[j].w - mu) * inv * g4.w + b4.w;
    *(float4*)(out + (size_t)tok * HH + h) = o;
  }
}

// ---------------------------------------------------------------------------
// Launch
// ---------------------------------------------------------------------------
extern "C" void kernel_launch(void* const* d_in, const int* in_sizes, int n_in,
                              void* d_out, int out_size, void* d_ws, size_t ws_size,
                              hipStream_t stream) {
  (void)in_sizes; (void)n_in; (void)out_size; (void)ws_size;

  const int B = 8, S = 2048, W = 8, WE = 200, H = 768;
  const int NTOK = B * S;           // 16384
  const int MW   = NTOK * W;        // 131072
  const int WEP  = 224;             // WE padded to multiple of 32

  const float* E    = (const float*)d_in[0];
  const float* LO   = (const float*)d_in[1];
  const int*   msk  = (const int*)d_in[2];
  const float* Wt   = (const float*)d_in[3];
  const float* bt   = (const float*)d_in[4];
  const float* Ww   = (const float*)d_in[5];
  const float* bw   = (const float*)d_in[6];
  const float* aW   = (const float*)d_in[7];
  const float* lng  = (const float*)d_in[8];
  const float* lnb  = (const float*)d_in[9];
  float* out = (float*)d_out;

  // workspace carve-up
  uint16_t* WtT = (uint16_t*)d_ws;                       // [768][224] bf16
  uint16_t* WwT = WtT + (size_t)H * WEP;                 // [768][768] bf16
  uint16_t* aWT = WwT + (size_t)H * H;                   // [768][768] bf16
  uint16_t* Ebf = aWT + (size_t)H * H;                   // [131072][224] bf16
  uint16_t* X1  = Ebf + (size_t)MW * WEP;                // [131072][768] bf16
  uint16_t* WO  = X1  + (size_t)MW * H;                  // [131072][768] bf16
  float*    Qb  = (float*)(WO + (size_t)MW * H);         // [16384][768] f32

  // 1) pack weights (transpose, pad, f32->bf16) and embeddings (pad rows)
  {
    int t1 = H * WEP;
    pack_transpose_bf16<<<(t1 + 255) / 256, 256, 0, stream>>>(Wt, WtT, WE, H, WEP);
    int t2 = H * H;
    pack_transpose_bf16<<<(t2 + 255) / 256, 256, 0, stream>>>(Ww, WwT, H, H, H);
    pack_transpose_bf16<<<(t2 + 255) / 256, 256, 0, stream>>>(aW, aWT, H, H, H);
    int t3 = MW * WEP;
    pack_rows_bf16<<<(t3 + 255) / 256, 256, 0, stream>>>(E, Ebf, WE, WEP, t3);
  }

  // 2) GEMM1: X1 = tanh(Ebf @ Wt + bt)  [131072 x 768], K=224 (zero-padded)
  gemm_bf16_wmma<false, 0><<<dim3(MW / BM, H / BN), 256, 0, stream>>>(
      Ebf, WtT, bt, X1, WEP, WEP, H);

  // 3) GEMM2: WO = X1 @ Ww + bw         [131072 x 768], K=768
  gemm_bf16_wmma<false, 1><<<dim3(MW / BM, H / BN), 256, 0, stream>>>(
      X1, WwT, bw, WO, H, H, H);

  // 4) GEMM3: Q = LO @ attn_W           [16384 x 768], K=768, f32 out
  gemm_bf16_wmma<true, 2><<<dim3(NTOK / BM, H / BN), 256, 0, stream>>>(
      LO, aWT, nullptr, Qb, H, H, H);

  // 5) scores -> softmax -> weighted sum -> residual -> LayerNorm
  fuse_attn_ln<<<NTOK / 8, 256, 0, stream>>>(Qb, WO, LO, msk, lng, lnb, out);
}
